// CoLightGAT_78426102825165
// MI455X (gfx1250) — matrix operand order repaired and checked by reading, not compiled, not run on hardware
//
#include <hip/hip_runtime.h>
#include <hip/hip_bf16.h>
#include <math.h>

// ---------------- problem constants (match reference) ----------------
#define B_TOTAL 16384
#define OBS     64
#define ACT     8
#define H       128
#define NNODE   5      // self + 4 neighbors
#define SLOPE   0.2f

#define WAVES_PER_WG 2
#define TILE_M       16
#define NTILES       (B_TOTAL / TILE_M)   // 1024

// padded LDS row strides (halves) to avoid bank conflicts on per-row A loads
#define LDH 130        // 128-wide activation rows (260B stride -> distinct banks)
#define LDF 260        // 256-wide "final" rows

// ---- f16-transposed weight layout inside d_ws (units: halves) --------
// Each matrix stored as WT[n][k] so a B fragment reads contiguous k.
#define OFS_W1 0                      // [128][64]
#define OFS_W2 8192                   // [128][128]
#define OFS_GW 24576                  // 2 x [128][128] (per head)
#define OFS_VW 57344                  // [128][256]
#define OFS_PW 90112                  // [128][256]
#define WS_HALVES 122880
#define WS_BYTES  (WS_HALVES * 2)     // 245760

typedef _Float16 half_t;
typedef __attribute__((ext_vector_type(16))) _Float16 v16h;
typedef __attribute__((ext_vector_type(8)))  float    v8f;

union FragU { v16h v; unsigned int u[8]; };
union H2U  { struct { half_t lo, hi; } h; unsigned int u; };

__device__ __forceinline__ void lds_fence() {
  // LDS ops from one wave are processed in order; after dscnt==0 all our
  // prior ds_stores are visible to our subsequent ds_loads (cross-lane).
  asm volatile("s_wait_dscnt 0" ::: "memory");
}

// ---- A fragment, f16 16x32 (MxK). lanes 0-15 row m=lane, K in 16-chunks
// across lane halves: k = kbase + 16*(lane>=16) + 2v, pair (k,k+1).
__device__ __forceinline__ v16h fragA_lds(const half_t* rowptr, int kbase) {
  const int l   = threadIdx.x & 31;
  const int khi = (l >> 4) << 4;
  FragU f;
#pragma unroll
  for (int v = 0; v < 8; ++v) {
    int k = kbase + khi + 2 * v;
    f.u[v] = *(const unsigned int*)(rowptr + k);
  }
  return f.v;
}

__device__ __forceinline__ v16h fragA_global_f32(const float* rowptr, int kbase) {
  const int l   = threadIdx.x & 31;
  const int khi = (l >> 4) << 4;
  FragU f;
#pragma unroll
  for (int v = 0; v < 8; ++v) {
    int k = kbase + khi + 2 * v;
    H2U p; p.h.lo = (half_t)rowptr[k]; p.h.hi = (half_t)rowptr[k + 1];
    f.u[v] = p.u;
  }
  return f.v;
}

// ---- B fragment, f16 32x16 (KxN). Same k order as fragA_lds (not the ISA table order): K in
// 16-chunks across lane halves: lane holds col n=lane&15,
// k = kbase + 16*(lane>=16) + 2v, pair (k,k+1).
__device__ __forceinline__ v16h fragB_lds(const half_t* colptr, int kbase) {
  const int l   = threadIdx.x & 31;
  const int khi = (l >> 4) << 4;
  FragU f;
#pragma unroll
  for (int v = 0; v < 8; ++v) {
    int k = kbase + khi + 2 * v;
    f.u[v] = *(const unsigned int*)(colptr + k);
  }
  return f.v;
}

// B fragment from pre-transposed f16 weights WT[n][ldk]: contiguous 32B per
// lane -> compiler merges into two global_load_b128, no cvt VALU.
__device__ __forceinline__ v16h fragB_g16(const half_t* WT, int ldk,
                                          int kbase, int nbase) {
  const int l   = threadIdx.x & 31;
  const int n   = nbase + (l & 15);
  const int khi = (l >> 4) << 4;
  const half_t* rp = WT + (size_t)n * ldk + kbase + khi;
  FragU f;
#pragma unroll
  for (int v = 0; v < 8; ++v) f.u[v] = *(const unsigned int*)(rp + 2 * v);
  return f.v;
}

// Fallback: B fragment straight from row-major f32 weight W[K][N].
__device__ __forceinline__ v16h fragB_g32(const float* W, int ldN,
                                          int kbase, int nbase) {
  const int l   = threadIdx.x & 31;
  const int n   = nbase + (l & 15);
  const int khi = (l >> 4) << 4;
  FragU f;
#pragma unroll
  for (int v = 0; v < 8; ++v) {
    int k = kbase + khi + 2 * v;
    H2U p; p.h.lo = (half_t)W[k * ldN + n]; p.h.hi = (half_t)W[(k + 1) * ldN + n];
    f.u[v] = p.u;
  }
  return f.v;
}

// ---- C tile (f32 16x16) store to f16 LDS. lane: col n = lane&15;
// VGPR r -> row m = r + 8*(lane>=16). Separate variants: no runtime branches.
__device__ __forceinline__ void storeC_plain(half_t* dst, int ld, int nbase, const v8f c) {
  const int l = threadIdx.x & 31, n = nbase + (l & 15), mofs = (l >> 4) << 3;
#pragma unroll
  for (int r = 0; r < 8; ++r) dst[(mofs + r) * ld + n] = (half_t)c[r];
}
__device__ __forceinline__ void storeC_bias(half_t* dst, int ld, int nbase,
                                            const v8f c, const float* bias) {
  const int l = threadIdx.x & 31, n = nbase + (l & 15), mofs = (l >> 4) << 3;
  const float bv = bias[n];
#pragma unroll
  for (int r = 0; r < 8; ++r) dst[(mofs + r) * ld + n] = (half_t)(c[r] + bv);
}
__device__ __forceinline__ void storeC_bias_relu(half_t* dst, int ld, int nbase,
                                                 const v8f c, const float* bias) {
  const int l = threadIdx.x & 31, n = nbase + (l & 15), mofs = (l >> 4) << 3;
  const float bv = bias[n];
#pragma unroll
  for (int r = 0; r < 8; ++r)
    dst[(mofs + r) * ld + n] = (half_t)fmaxf(c[r] + bv, 0.0f);
}

#define WMMA(a, b, c) \
  __builtin_amdgcn_wmma_f32_16x16x32_f16(false, (a), false, (b), (short)0, (c), false, false)

#define V8F_ZERO {0.f,0.f,0.f,0.f,0.f,0.f,0.f,0.f}

struct WaveScratch {
  half_t emb[NNODE * 16 * LDH];  // emb[node][b][d]; reused later: final[16][LDF] + hv + hp
  half_t S[32 * LDH];            // h1[16][LDH] during encoder; ctx[2][16][LDH] later
  float  escr[NNODE * 16 * 4];   // per node, per b: {ei_h0, ej_h0, ei_h1, ej_h1}
};

// =======================================================================
// prep kernel: convert the 5 weight matrices to f16, transposed [n][k]
// =======================================================================
__global__ void prep_weights(const float* __restrict__ enc_w1,
                             const float* __restrict__ enc_w2,
                             const float* __restrict__ gat_w,
                             const float* __restrict__ val_w1,
                             const float* __restrict__ pol_w1,
                             half_t* __restrict__ ws) {
  for (int e = blockIdx.x * blockDim.x + threadIdx.x; e < WS_HALVES;
       e += gridDim.x * blockDim.x) {
    float v;
    if (e < OFS_W2) {                       // enc_w1T [128][64]
      int i = e - OFS_W1, n = i >> 6, k = i & 63;
      v = enc_w1[k * H + n];
    } else if (e < OFS_GW) {                // enc_w2T [128][128]
      int i = e - OFS_W2, n = i >> 7, k = i & 127;
      v = enc_w2[k * H + n];
    } else if (e < OFS_VW) {                // gat_wT  2 x [128][128]
      int i = e - OFS_GW, h = i >> 14, j = i & 16383, n = j >> 7, k = j & 127;
      v = gat_w[(h * H + k) * H + n];
    } else if (e < OFS_PW) {                // val_w1T [128][256]
      int i = e - OFS_VW, n = i >> 8, k = i & 255;
      v = val_w1[k * H + n];
    } else {                                // pol_w1T [128][256]
      int i = e - OFS_PW, n = i >> 8, k = i & 255;
      v = pol_w1[k * H + n];
    }
    ws[e] = (half_t)v;
  }
}

// =======================================================================
// main kernel (templated: F16W = use pre-transposed f16 weights in ws)
// =======================================================================
template <bool F16W>
__global__ __launch_bounds__(WAVES_PER_WG * 32)
void colight_gat_kernel(const float* __restrict__ obs,
                        const float* __restrict__ nobs,
                        const int*   __restrict__ adj,
                        const float* __restrict__ enc_w1, const float* __restrict__ enc_b1,
                        const float* __restrict__ enc_w2, const float* __restrict__ enc_b2,
                        const float* __restrict__ gat_w,  const float* __restrict__ gat_a,
                        const float* __restrict__ val_w1, const float* __restrict__ val_b1,
                        const float* __restrict__ val_w2, const float* __restrict__ val_b2,
                        const float* __restrict__ pol_w1, const float* __restrict__ pol_b1,
                        const float* __restrict__ pol_w2, const float* __restrict__ pol_b2,
                        const half_t* __restrict__ wsw,
                        float* __restrict__ out, int ntiles)
{
  __shared__ WaveScratch wsc[WAVES_PER_WG];
  __shared__ half_t Ba[16 * LDH];  // ei/ej B: cols {wa1_h0, wa2_h0, wa1_h1, wa2_h1, 0..}

  const int tid = threadIdx.x;

  // ---- cooperative precompute: wa[c][d] = sum_o gat_w[h][d][o] * a_sel[h][o]
  for (int e = tid; e < 4 * H; e += WAVES_PER_WG * 32) {
    int c = e >> 7, d = e & (H - 1);
    int h = c >> 1, sel = c & 1;
    const float* grow = gat_w + (h * H + d) * H;
    const float* avec = gat_a + h * 2 * H + sel * H;
    float acc = 0.0f;
    for (int o = 0; o < H; ++o) acc += grow[o] * avec[o];
    Ba[c * LDH + d] = (half_t)acc;
  }
  for (int e = tid; e < 12 * H; e += WAVES_PER_WG * 32) {
    Ba[(4 + (e >> 7)) * LDH + (e & (H - 1))] = (half_t)0.0f;
  }
  __syncthreads();

  const int wv = tid >> 5;
  const int l  = tid & 31;
  WaveScratch& W = wsc[wv];
  const int tile = blockIdx.x * WAVES_PER_WG + wv;
  if (tile >= ntiles) return;                 // wave-uniform

  const int brow = tile * TILE_M + (l & 15);  // this lane's batch row

  // weight-fragment dispatch (dead branch eliminated at compile time)
#define FB_W1(ks, nb) (F16W ? fragB_g16(wsw + OFS_W1, OBS, (ks), (nb)) \
                            : fragB_g32(enc_w1, H, (ks), (nb)))
#define FB_W2(ks, nb) (F16W ? fragB_g16(wsw + OFS_W2, H, (ks), (nb)) \
                            : fragB_g32(enc_w2, H, (ks), (nb)))
#define FB_GW(h, ks, nb) (F16W ? fragB_g16(wsw + OFS_GW + (h) * H * H, H, (ks), (nb)) \
                               : fragB_g32(gat_w + (h) * H * H, H, (ks), (nb)))
#define FB_VW(ks, nb) (F16W ? fragB_g16(wsw + OFS_VW, 2 * H, (ks), (nb)) \
                            : fragB_g32(val_w1, H, (ks), (nb)))
#define FB_PW(ks, nb) (F16W ? fragB_g16(wsw + OFS_PW, 2 * H, (ks), (nb)) \
                            : fragB_g32(pol_w1, H, (ks), (nb)))

  // ================= encoder + ei/ej per node tile =================
  for (int t = 0; t < NNODE; ++t) {
    const float* xrow = (t == 0) ? (obs + (size_t)brow * OBS)
                                 : (nobs + ((size_t)brow * (NNODE - 1) + (t - 1)) * OBS);
    v16h xa0 = fragA_global_f32(xrow, 0);
    v16h xa1 = fragA_global_f32(xrow, 32);

    // h1 = relu(X @ enc_w1 + b1) -> W.S[16][LDH]
#pragma unroll
    for (int c = 0; c < 8; ++c) {
      v8f acc = V8F_ZERO;
      acc = WMMA(xa0, FB_W1(0,  c * 16), acc);
      acc = WMMA(xa1, FB_W1(32, c * 16), acc);
      storeC_bias_relu(W.S, LDH, c * 16, acc, enc_b1);
    }
    lds_fence();

    // emb = h1 @ enc_w2 + b2 -> W.emb[t]
    {
      v16h ha[4];
#pragma unroll
      for (int ks = 0; ks < 4; ++ks) ha[ks] = fragA_lds(W.S + (l & 15) * LDH, ks * 32);
#pragma unroll
      for (int c = 0; c < 8; ++c) {
        v8f acc = V8F_ZERO;
#pragma unroll
        for (int ks = 0; ks < 4; ++ks) acc = WMMA(ha[ks], FB_W2(ks * 32, c * 16), acc);
        storeC_bias(W.emb + t * 16 * LDH, LDH, c * 16, acc, enc_b2);
      }
    }
    lds_fence();

    // ei/ej = emb_tile @ [wa1|wa2 per head] : one C tile, cols 0..3 used
    {
      v8f eacc = V8F_ZERO;
#pragma unroll
      for (int ks = 0; ks < 4; ++ks) {
        v16h ea = fragA_lds(W.emb + (t * 16 + (l & 15)) * LDH, ks * 32);
        eacc = WMMA(ea, fragB_lds(Ba + (l & 15) * LDH, ks * 32), eacc);
      }
      const int n = l & 15, mofs = (l >> 4) << 3;
      if (n < 4) {
#pragma unroll
        for (int r = 0; r < 8; ++r)
          W.escr[(t * 16 + mofs + r) * 4 + n] = eacc[r];
      }
    }
    lds_fence();
  }

  // ========= attention (row i=0 only) + context combine =========
  {
    const int b  = l & 15;
    const int bg = tile * TILE_M + b;
    int ad[NNODE];
#pragma unroll
    for (int j = 0; j < NNODE; ++j) ad[j] = adj[bg * NNODE + j];

    float at[2][NNODE];
#pragma unroll
    for (int h = 0; h < 2; ++h) {
      const float ei = W.escr[(0 * 16 + b) * 4 + h * 2 + 0];  // node 0 = self
      float ev[NNODE];
      float mx = -1e30f;
#pragma unroll
      for (int j = 0; j < NNODE; ++j) {
        float e = ei + W.escr[(j * 16 + b) * 4 + h * 2 + 1];
        e = (e > 0.0f) ? e : SLOPE * e;                       // leaky_relu
        if (ad[j] == 0) e = -1e30f;                           // mask (adj[:,0]==1 always)
        ev[j] = e;
        mx = fmaxf(mx, e);
      }
      float s = 0.0f;
#pragma unroll
      for (int j = 0; j < NNODE; ++j) { ev[j] = __expf(ev[j] - mx); s += ev[j]; }
      const float inv = 1.0f / s;
#pragma unroll
      for (int j = 0; j < NNODE; ++j) at[h][j] = ev[j] * inv;
    }
    // ctx[h][b][d] = sum_j attn[h][j] * emb[j][b][d]
    const int ohalf = (l >> 4) * 64;
#pragma unroll
    for (int h = 0; h < 2; ++h) {
      for (int o = 0; o < 64; ++o) {
        const int oo = ohalf + o;
        float acc = 0.0f;
#pragma unroll
        for (int j = 0; j < NNODE; ++j)
          acc += at[h][j] * (float)W.emb[(j * 16 + b) * LDH + oo];
        W.S[(h * 16 + b) * LDH + oo] = (half_t)acc;
      }
    }
  }
  lds_fence();

  // ========= final = [ctx0 @ gat_w0 | ctx1 @ gat_w1] -> W.emb[16][LDF]
  half_t* finalb = W.emb;
#pragma unroll
  for (int h = 0; h < 2; ++h) {
    v16h ca[4];
#pragma unroll
    for (int ks = 0; ks < 4; ++ks)
      ca[ks] = fragA_lds(W.S + (h * 16 + (l & 15)) * LDH, ks * 32);
#pragma unroll
    for (int c = 0; c < 8; ++c) {
      v8f acc = V8F_ZERO;
#pragma unroll
      for (int ks = 0; ks < 4; ++ks) acc = WMMA(ca[ks], FB_GW(h, ks * 32, c * 16), acc);
      storeC_plain(finalb + h * H, LDF, c * 16, acc);
    }
  }
  lds_fence();

  // ========= value / policy hidden layers (K=256) =========
  half_t* hv = W.emb + 16 * LDF;
  half_t* hp = hv + 16 * LDH;
  {
    v16h fa[8];
#pragma unroll
    for (int ks = 0; ks < 8; ++ks)
      fa[ks] = fragA_lds(finalb + (l & 15) * LDF, ks * 32);
#pragma unroll
    for (int c = 0; c < 8; ++c) {
      v8f av = V8F_ZERO, ap = V8F_ZERO;
#pragma unroll
      for (int ks = 0; ks < 8; ++ks) {
        av = WMMA(fa[ks], FB_VW(ks * 32, c * 16), av);
        ap = WMMA(fa[ks], FB_PW(ks * 32, c * 16), ap);
      }
      storeC_bias_relu(hv, LDH, c * 16, av, val_b1);
      storeC_bias_relu(hp, LDH, c * 16, ap, pol_b1);
    }
  }
  lds_fence();

  // ========= tiny projections (128->1 and 128->8) in VALU =========
  {
    const int b  = l & 15;
    const int bg = tile * TILE_M + b;
    if (l < 16) {
      float acc = val_b2[0];
      for (int d = 0; d < H; ++d) acc += (float)hv[b * LDH + d] * val_w2[d];
      out[bg] = acc;                              // value output
    }
    const int a0 = (l >> 4) * 4;
#pragma unroll
    for (int ai = 0; ai < 4; ++ai) {
      const int a = a0 + ai;
      float acc = pol_b2[a];
      for (int d = 0; d < H; ++d) acc += (float)hp[b * LDH + d] * pol_w2[d * ACT + a];
      out[B_TOTAL + (size_t)bg * ACT + a] = acc;  // logits output
    }
  }
#undef FB_W1
#undef FB_W2
#undef FB_GW
#undef FB_VW
#undef FB_PW
}

extern "C" void kernel_launch(void* const* d_in, const int* in_sizes, int n_in,
                              void* d_out, int out_size, void* d_ws, size_t ws_size,
                              hipStream_t stream) {
  const float* obs    = (const float*)d_in[0];
  const float* nobs   = (const float*)d_in[1];
  const int*   adj    = (const int*)  d_in[2];
  const float* enc_w1 = (const float*)d_in[3];
  const float* enc_b1 = (const float*)d_in[4];
  const float* enc_w2 = (const float*)d_in[5];
  const float* enc_b2 = (const float*)d_in[6];
  const float* gat_w  = (const float*)d_in[7];
  const float* gat_a  = (const float*)d_in[8];
  const float* val_w1 = (const float*)d_in[9];
  const float* val_b1 = (const float*)d_in[10];
  const float* val_w2 = (const float*)d_in[11];
  const float* val_b2 = (const float*)d_in[12];
  const float* pol_w1 = (const float*)d_in[13];
  const float* pol_b1 = (const float*)d_in[14];
  const float* pol_w2 = (const float*)d_in[15];
  const float* pol_b2 = (const float*)d_in[16];
  float* out = (float*)d_out;

  const int ntiles = NTILES;                                       // 1024
  const int blocks = (ntiles + WAVES_PER_WG - 1) / WAVES_PER_WG;   // 512

  const bool use_f16w = (d_ws != nullptr) && (ws_size >= (size_t)WS_BYTES);
  if (use_f16w) {
    half_t* wsw = (half_t*)d_ws;
    prep_weights<<<480, 256, 0, stream>>>(enc_w1, enc_w2, gat_w, val_w1, pol_w1, wsw);
    colight_gat_kernel<true><<<blocks, WAVES_PER_WG * 32, 0, stream>>>(
        obs, nobs, adj, enc_w1, enc_b1, enc_w2, enc_b2, gat_w, gat_a,
        val_w1, val_b1, val_w2, val_b2, pol_w1, pol_b1, pol_w2, pol_b2,
        wsw, out, ntiles);
  } else {
    colight_gat_kernel<false><<<blocks, WAVES_PER_WG * 32, 0, stream>>>(
        obs, nobs, adj, enc_w1, enc_b1, enc_w2, enc_b2, gat_w, gat_a,
        val_w1, val_b1, val_w2, val_b2, pol_w1, pol_b1, pol_w2, pol_b2,
        nullptr, out, ntiles);
  }
}
